// CausalSelfAttention_84928683311105
// MI455X (gfx1250) — matrix operand order repaired and checked by reading, not compiled, not run on hardware
//
#include <hip/hip_runtime.h>
#include <hip/hip_bf16.h>

typedef __bf16 bf16;
typedef bf16  bf16x8  __attribute__((ext_vector_type(8)));
typedef bf16  v16bf   __attribute__((ext_vector_type(16)));
typedef float v8f     __attribute__((ext_vector_type(8)));

#define Bq   2
#define Tq   2048
#define Cq   2048
#define NH   16
#define NKV  4
#define HD   128

// ---- CDNA5 async global->LDS copy helpers (ASYNCcnt-tracked, no VGPR bounce) ----
static __device__ inline unsigned lds_off(void* p) {
  // generic -> addrspace(3) -> raw LDS byte offset
  return (unsigned)(unsigned long long)(__attribute__((address_space(3))) void*)p;
}
static __device__ inline void async_copy_b128(void* ldsDst, const void* gSrc) {
  unsigned l = lds_off(ldsDst);
  asm volatile("global_load_async_to_lds_b128 %0, %1, off"
               :: "v"(l), "v"(gSrc)
               : "memory");
}
static __device__ inline void wait_async0(void) {
  asm volatile("s_wait_asynccnt 0x0" ::: "memory");
}

static __device__ inline v16bf make16(bf16x8 lo, bf16x8 hi) {
  return __builtin_shufflevector(lo, hi, 0,1,2,3,4,5,6,7,8,9,10,11,12,13,14,15);
}

// A-fragment (16-bit A, 16x32): lanes 0-15 hold K {0..7, 16..23}, lanes 16-31 hold K {8..15, 24..31}
static __device__ inline v16bf ldsA(const bf16* p, int row, int ld, int kc, int lane) {
  int kb = kc * 32 + ((lane & 16) ? 8 : 0);
  bf16x8 lo = *(const bf16x8*)(p + row * ld + kb);
  bf16x8 hi = *(const bf16x8*)(p + row * ld + kb + 16);
  return make16(lo, hi);
}

// B-fragment (16-bit B, 32x16): lanes 0-15 hold K {0..7, 16..23}, lanes 16-31 hold K {8..15, 24..31}
// p is stored column-major-in-N: p[col*ld + k] (K contiguous per output column)
static __device__ inline v16bf ldsB(const bf16* p, int col, int ld, int kc, int lane) {
  int kb = kc * 32 + ((lane & 16) ? 8 : 0);
  bf16x8 lo = *(const bf16x8*)(p + col * ld + kb);
  bf16x8 hi = *(const bf16x8*)(p + col * ld + kb + 16);
  return make16(lo, hi);
}

static __device__ inline v8f wmma_bf16(v16bf a, v16bf b, v8f c) {
  return __builtin_amdgcn_wmma_f32_16x16x32_bf16(false, a, false, b, (short)0, c, false, false);
}

// ---------------- fp32 -> bf16 converts ----------------
__global__ __launch_bounds__(256) void f32_to_bf16(const float* __restrict__ src,
                                                   bf16* __restrict__ dst, size_t n) {
  size_t i = (size_t)blockIdx.x * blockDim.x + threadIdx.x;
  if (i < n) dst[i] = (bf16)src[i];
}

__global__ __launch_bounds__(256) void f32_to_bf16_strided(const float* __restrict__ src,
                                                           bf16* __restrict__ dst,
                                                           int rows, int cols, int dstStride) {
  size_t i = (size_t)blockIdx.x * blockDim.x + threadIdx.x;
  size_t n = (size_t)rows * cols;
  if (i < n) {
    int r = (int)(i / cols), c = (int)(i % cols);
    dst[(size_t)r * dstStride + c] = (bf16)src[i];
  }
}

// ---------------- bf16 WMMA GEMM: C(f32)[M,N] = A[M,K] * B[K,N] ----------------
__global__ __launch_bounds__(256) void gemm_bf16(const bf16* __restrict__ A,
                                                 const bf16* __restrict__ Bm,
                                                 float* __restrict__ C,
                                                 int M, int N, int K) {
  __shared__ bf16 As[128 * 32];   // [row][k]
  __shared__ bf16 Bs[128 * 32];   // [n][k]  (K-transposed)

  const int tid  = threadIdx.x;
  const int lane = tid & 31;
  const int wave = tid >> 5;
  const int wm = wave >> 1;       // 0..3 -> 32-row slice
  const int wn = wave & 1;        // 0..1 -> 64-col slice
  const int nl = lane & 15;
  const int half8 = (lane & 16) ? 8 : 0;
  const int bm = blockIdx.y * 128;
  const int bn = blockIdx.x * 128;

  v8f acc[2][4];
  #pragma unroll
  for (int i = 0; i < 2; ++i)
    #pragma unroll
    for (int j = 0; j < 4; ++j) acc[i][j] = (v8f)0.0f;

  for (int k0 = 0; k0 < K; k0 += 32) {
    // stage A tile: 128x32 via async global->LDS (2 x b128 per thread)
    {
      int row = tid >> 1, kh = (tid & 1) * 16;
      const bf16* g = A + (size_t)(bm + row) * K + k0 + kh;
      bf16* s = As + row * 32 + kh;
      async_copy_b128(s,     g);
      async_copy_b128(s + 8, g + 8);
      if (k0 + 32 < K)
        __builtin_prefetch(g + 32, 0, 1);
    }
    // stage B tile transposed: thread owns one k, 16 columns (VGPR bounce needed)
    {
      int kk = tid >> 3, nc = (tid & 7) * 16;
      const bf16* g = Bm + (size_t)(k0 + kk) * N + bn + nc;
      bf16 tmp[16];
      *(uint4*)(tmp + 0) = *(const uint4*)(g + 0);
      *(uint4*)(tmp + 8) = *(const uint4*)(g + 8);
      #pragma unroll
      for (int j = 0; j < 16; ++j) Bs[(nc + j) * 32 + kk] = tmp[j];
      if (k0 + 32 < K)
        __builtin_prefetch(Bm + (size_t)(k0 + 32 + kk) * N + bn + nc, 0, 1);
    }
    wait_async0();
    __syncthreads();

    v16bf a[2], b[4];
    #pragma unroll
    for (int mi = 0; mi < 2; ++mi) a[mi] = ldsA(As, wm * 32 + mi * 16 + nl, 32, 0, lane);
    #pragma unroll
    for (int ni = 0; ni < 4; ++ni) b[ni] = ldsB(Bs, wn * 64 + ni * 16 + nl, 32, 0, lane);
    #pragma unroll
    for (int mi = 0; mi < 2; ++mi)
      #pragma unroll
      for (int ni = 0; ni < 4; ++ni)
        acc[mi][ni] = wmma_bf16(a[mi], b[ni], acc[mi][ni]);
    __syncthreads();
  }

  #pragma unroll
  for (int mi = 0; mi < 2; ++mi)
    #pragma unroll
    for (int ni = 0; ni < 4; ++ni)
      #pragma unroll
      for (int j = 0; j < 8; ++j)
        C[(size_t)(bm + wm * 32 + mi * 16 + j + half8) * N + bn + wn * 64 + ni * 16 + nl] =
            acc[mi][ni][j];
}

// ---------------- RoPE + RMS-norm + bf16 pack (one wave per (b,t,channel)) ----------------
// channels: 0..15 = q heads, 16..19 = k heads, 20..23 = v heads
__global__ __launch_bounds__(256) void rope_rms(const float* __restrict__ qkv,
                                                const float* __restrict__ cosb,
                                                const float* __restrict__ sinb,
                                                bf16* __restrict__ qb,   // [B*NH][T][HD]
                                                bf16* __restrict__ kb,   // [B*NKV][T][HD]
                                                bf16* __restrict__ vtb)  // [B*NKV][HD][T]
{
  const int lane = threadIdx.x & 31;
  const int gw = blockIdx.x * 8 + (threadIdx.x >> 5);
  const int bt = gw / 24;
  const int ch = gw % 24;
  const int b = bt >> 11;           // /T
  const int t = bt & (Tq - 1);

  const float* row = qkv + (size_t)bt * 3072;
  int base;
  if (ch < 16)       base = ch * HD;
  else if (ch < 20)  base = 2048 + (ch - 16) * HD;
  else               base = 2560 + (ch - 20) * HD;

  float x0 = row[base + lane];
  float x1 = row[base + lane + 32];
  float x2 = row[base + lane + 64];
  float x3 = row[base + lane + 96];

  if (ch < 20) {
    // rotary pairs (d, d+64): (lane, lane+64) and (lane+32, lane+96)
    float c0 = cosb[t * 64 + lane],      s0 = sinb[t * 64 + lane];
    float c1 = cosb[t * 64 + lane + 32], s1 = sinb[t * 64 + lane + 32];
    float r0 = x0 * c0 - x2 * s0;
    float r2 = x0 * s0 + x2 * c0;
    float r1 = x1 * c1 - x3 * s1;
    float r3 = x1 * s1 + x3 * c1;
    float ss = r0 * r0 + r1 * r1 + r2 * r2 + r3 * r3;
    #pragma unroll
    for (int off = 16; off > 0; off >>= 1) ss += __shfl_xor(ss, off);
    float scale = __frsqrt_rn(ss * (1.0f / HD) + 1e-6f);
    if (ch < 16) scale *= (1.0f / 128.0f);   // fold both 1/sqrt(HD) factors into q
    bf16* dst = (ch < 16)
        ? qb + ((size_t)(b * NH + ch) * Tq + t) * HD
        : kb + ((size_t)(b * NKV + (ch - 16)) * Tq + t) * HD;
    dst[lane]      = (bf16)(r0 * scale);
    dst[lane + 32] = (bf16)(r1 * scale);
    dst[lane + 64] = (bf16)(r2 * scale);
    dst[lane + 96] = (bf16)(r3 * scale);
  } else {
    bf16* dst = vtb + (size_t)(b * NKV + (ch - 20)) * HD * Tq;
    dst[(size_t)(lane)      * Tq + t] = (bf16)x0;
    dst[(size_t)(lane + 32) * Tq + t] = (bf16)x1;
    dst[(size_t)(lane + 64) * Tq + t] = (bf16)x2;
    dst[(size_t)(lane + 96) * Tq + t] = (bf16)x3;
  }
}

// ---------------- Flash attention (WMMA, online softmax) ----------------
// grid.x = B*NH, grid.y = T/128 ; block = 256 ; dyn LDS = 4 * 32KB
__global__ __launch_bounds__(256) void attn_kernel(const bf16* __restrict__ qb,
                                                   const bf16* __restrict__ kb,
                                                   const bf16* __restrict__ vtb,
                                                   bf16* __restrict__ yb) {
  extern __shared__ char smem[];
  bf16* Qs = (bf16*)smem;            // [128][128] row=q, col=d
  bf16* Ks = Qs + 128 * 128;         // [128][128] row=s, col=d
  bf16* Vs = Ks + 128 * 128;         // [128][128] row=d, col=s
  bf16* Ps = Vs + 128 * 128;         // [8 waves][16][128]

  const int tid  = threadIdx.x;
  const int lane = tid & 31;
  const int wave = tid >> 5;
  const int nl = lane & 15;
  const int half8 = (lane & 16) ? 8 : 0;
  const int rbase = wave * 16;

  const int b  = blockIdx.x >> 4;
  const int hq = blockIdx.x & 15;
  const int hk = hq >> 2;
  const int qt = blockIdx.y;
  const int qbase = qt * 128;

  const bf16* qg = qb + ((size_t)(b * NH + hq) * Tq + qbase) * HD;
  const bf16* kg = kb + (size_t)(b * NKV + hk) * Tq * HD;
  const bf16* vg = vtb + (size_t)(b * NKV + hk) * HD * Tq;

  // stage Q tile (async global->LDS, 8 x b128 per thread)
  {
    int row = tid >> 1, c0 = (tid & 1) * 64;
    const bf16* g = qg + (size_t)row * HD + c0;
    bf16* s = Qs + row * 128 + c0;
    #pragma unroll
    for (int i = 0; i < 8; ++i) async_copy_b128(s + i * 8, g + i * 8);
  }
  wait_async0();
  __syncthreads();

  v16bf qa[4];
  #pragma unroll
  for (int kc = 0; kc < 4; ++kc) qa[kc] = ldsA(Qs, rbase + nl, 128, kc, lane);

  v8f O[8];
  float m[8], l[8];
  #pragma unroll
  for (int j = 0; j < 8; ++j) { O[j] = (v8f)0.0f; m[j] = -__builtin_inff(); l[j] = 0.0f; }

  for (int st = 0; st <= qt; ++st) {
    __syncthreads();   // previous round done before overwriting K/V tiles
    {
      int row = tid >> 1, c0 = (tid & 1) * 64;
      const bf16* g = kg + ((size_t)(st * 128 + row)) * HD + c0;
      bf16* s = Ks + row * 128 + c0;
      #pragma unroll
      for (int i = 0; i < 8; ++i) async_copy_b128(s + i * 8, g + i * 8);
    }
    {
      int d = tid >> 1, c0 = (tid & 1) * 64;
      const bf16* g = vg + (size_t)d * Tq + st * 128 + c0;
      bf16* s = Vs + d * 128 + c0;
      #pragma unroll
      for (int i = 0; i < 8; ++i) async_copy_b128(s + i * 8, g + i * 8);
    }
    wait_async0();
    __syncthreads();

    // S = Q K^T (16 rows x 128 cols per wave)
    v8f S[8];
    #pragma unroll
    for (int ct = 0; ct < 8; ++ct) {
      S[ct] = (v8f)0.0f;
      #pragma unroll
      for (int kc = 0; kc < 4; ++kc)
        S[ct] = wmma_bf16(qa[kc], ldsB(Ks, ct * 16 + nl, 128, kc, lane), S[ct]);
    }

    if (st == qt) {   // causal mask on the diagonal tile
      #pragma unroll
      for (int ct = 0; ct < 8; ++ct) {
        int col = ct * 16 + nl;
        #pragma unroll
        for (int j = 0; j < 8; ++j)
          if (col > rbase + j + half8) S[ct][j] = -__builtin_inff();
      }
    }

    // online softmax: per-row (16-lane) max / sum
    float mnew[8];
    #pragma unroll
    for (int j = 0; j < 8; ++j) {
      mnew[j] = m[j];
      #pragma unroll
      for (int ct = 0; ct < 8; ++ct) mnew[j] = fmaxf(mnew[j], S[ct][j]);
      #pragma unroll
      for (int off = 8; off > 0; off >>= 1) mnew[j] = fmaxf(mnew[j], __shfl_xor(mnew[j], off));
    }

    float alpha[8], rs[8];
    #pragma unroll
    for (int j = 0; j < 8; ++j) { alpha[j] = __expf(m[j] - mnew[j]); rs[j] = 0.0f; m[j] = mnew[j]; }

    bf16* myP = Ps + (size_t)rbase * 128;
    #pragma unroll
    for (int ct = 0; ct < 8; ++ct)
      #pragma unroll
      for (int j = 0; j < 8; ++j) {
        float p = __expf(S[ct][j] - mnew[j]);
        rs[j] += p;
        myP[(j + half8) * 128 + ct * 16 + nl] = (bf16)p;
      }

    #pragma unroll
    for (int j = 0; j < 8; ++j) {
      #pragma unroll
      for (int off = 8; off > 0; off >>= 1) rs[j] += __shfl_xor(rs[j], off);
      l[j] = l[j] * alpha[j] + rs[j];
    }

    #pragma unroll
    for (int dt = 0; dt < 8; ++dt)
      #pragma unroll
      for (int j = 0; j < 8; ++j) O[dt][j] *= alpha[j];

    // O += P * V  (P staged per-wave in LDS -> A layout; Vs is [d][s] so B is contiguous)
    v16bf pa[4];
    #pragma unroll
    for (int kc = 0; kc < 4; ++kc) pa[kc] = ldsA(myP, nl, 128, kc, lane);
    #pragma unroll
    for (int dt = 0; dt < 8; ++dt)
      #pragma unroll
      for (int kc = 0; kc < 4; ++kc)
        O[dt] = wmma_bf16(pa[kc], ldsB(Vs, dt * 16 + nl, 128, kc, lane), O[dt]);
  }

  // epilogue: O /= l, write bf16 y[b][t][hq*HD + d]
  #pragma unroll
  for (int dt = 0; dt < 8; ++dt)
    #pragma unroll
    for (int j = 0; j < 8; ++j) {
      float v = O[dt][j] / l[j];
      yb[(size_t)(b * Tq + qbase + rbase + j + half8) * Cq + hq * HD + dt * 16 + nl] = (bf16)v;
    }
}

// ---------------- host launcher ----------------
extern "C" void kernel_launch(void* const* d_in, const int* in_sizes, int n_in,
                              void* d_out, int out_size, void* d_ws, size_t ws_size,
                              hipStream_t stream) {
  const float* x    = (const float*)d_in[0];
  const float* cosb = (const float*)d_in[1];
  const float* sinb = (const float*)d_in[2];
  const float* Wq   = (const float*)d_in[3];
  const float* Wk   = (const float*)d_in[4];
  const float* Wv   = (const float*)d_in[5];
  const float* Wp   = (const float*)d_in[6];
  (void)in_sizes; (void)n_in; (void)out_size; (void)ws_size;

  const size_t BT = (size_t)Bq * Tq;        // 4096
  char* ws = (char*)d_ws;
  size_t off = 0;
  bf16*  xb   = (bf16*)(ws + off); off += BT * Cq * 2;                 // 16 MB
  bf16*  wqkv = (bf16*)(ws + off); off += (size_t)Cq * 3072 * 2;       // 12 MB
  bf16*  wpB  = (bf16*)(ws + off); off += (size_t)Cq * Cq * 2;         // 8 MB
  float* qkv  = (float*)(ws + off); off += BT * 3072 * 4;              // 48 MB
  bf16*  qb   = (bf16*)(ws + off); off += (size_t)Bq * NH  * Tq * HD * 2;  // 16 MB
  bf16*  kbuf = (bf16*)(ws + off); off += (size_t)Bq * NKV * Tq * HD * 2;  // 4 MB
  bf16*  vtb  = (bf16*)(ws + off); off += (size_t)Bq * NKV * HD * Tq * 2;  // 4 MB
  bf16*  yb   = (bf16*)(ws + off); off += BT * Cq * 2;                 // 16 MB

  // 1) converts
  {
    size_t n = BT * Cq;
    f32_to_bf16<<<dim3((unsigned)((n + 255) / 256)), 256, 0, stream>>>(x, xb, n);
  }
  {
    size_t n = (size_t)Cq * Cq;
    f32_to_bf16_strided<<<dim3((unsigned)((n + 255) / 256)), 256, 0, stream>>>(Wq, wqkv, Cq, Cq, 3072);
  }
  {
    size_t n = (size_t)Cq * 512;
    f32_to_bf16_strided<<<dim3((unsigned)((n + 255) / 256)), 256, 0, stream>>>(Wk, wqkv + 2048, Cq, 512, 3072);
    f32_to_bf16_strided<<<dim3((unsigned)((n + 255) / 256)), 256, 0, stream>>>(Wv, wqkv + 2560, Cq, 512, 3072);
  }
  {
    size_t n = (size_t)Cq * Cq;
    f32_to_bf16<<<dim3((unsigned)((n + 255) / 256)), 256, 0, stream>>>(Wp, wpB, n);
  }

  // 2) fused QKV GEMM: [4096 x 2048] x [2048 x 3072]
  gemm_bf16<<<dim3(3072 / 128, (unsigned)(BT / 128)), 256, 0, stream>>>(xb, wqkv, qkv, (int)BT, 3072, Cq);

  // 3) RoPE + RMS-norm + packing (one wave per (b,t,channel); 24 channels)
  rope_rms<<<dim3((unsigned)(BT * 24 / 8)), 256, 0, stream>>>(qkv, cosb, sinb, qb, kbuf, vtb);

  // 4) flash attention
  (void)hipFuncSetAttribute((const void*)attn_kernel,
                            hipFuncAttributeMaxDynamicSharedMemorySize, 131072);
  attn_kernel<<<dim3(Bq * NH, Tq / 128), 256, 131072, stream>>>(qb, kbuf, vtb, yb);

  // 5) output projection: [4096 x 2048] x [2048 x 2048] -> d_out (fp32)
  gemm_bf16<<<dim3(Cq / 128, (unsigned)(BT / 128)), 256, 0, stream>>>(yb, wpB, (float*)d_out, (int)BT, Cq, Cq);
}
